// TemplatePairStack_15968688406916
// MI455X (gfx1250) — matrix operand order repaired and checked by reading, not compiled, not run on hardware
//
#include <hip/hip_runtime.h>
#include <hip/hip_bf16.h>
#include <cstdint>
#include <cstddef>

// ---------------------------------------------------------------------------
// Template pair stack for MI455X (gfx1250): all contractions on v_wmma
// f32_16x16x32_f16 (f16 in, f32 accum), wave32, 320KB-LDS WGP model.
// ---------------------------------------------------------------------------

#define TN 2            // N_TEMPL
#define RN 192          // N_RES
#define CD 64           // C_T == C_MUL
#define NH 4            // heads
#define ROWS (TN * RN * RN)   // 73728 pair rows

typedef _Float16 f16;
typedef __attribute__((ext_vector_type(16))) _Float16 v16h;
typedef __attribute__((ext_vector_type(8)))  float    v8f;

__device__ __forceinline__ float fastrcp(float x) {
#if __has_builtin(__builtin_amdgcn_rcpf)
  return __builtin_amdgcn_rcpf(x);        // v_rcp_f32
#else
  return 1.f / x;
#endif
}
__device__ __forceinline__ float fastrsq(float x) {
#if __has_builtin(__builtin_amdgcn_rsqf)
  return __builtin_amdgcn_rsqf(x);        // v_rsq_f32
#else
  return rsqrtf(x);
#endif
}
__device__ __forceinline__ float sigf(float x) {
  return fastrcp(1.f + __expf(-x));       // v_exp_f32 + v_rcp_f32
}

__device__ __forceinline__ v8f wmma_f16(v16h a, v16h b, v8f c) {
  return __builtin_amdgcn_wmma_f32_16x16x32_f16(false, a, false, b, (short)0, c,
                                                false, false);
}

// A fragment (16x32 f16): lane = row (lane&15); element e -> K = kb+16*hi+e
// (contiguous 16 halves).
__device__ __forceinline__ v16h load_a(const f16* __restrict__ X, int ld,
                                       int rowbase, int kb, int lane) {
  int m = lane & 15, hi = lane >> 4;
  const f16* r = X + (size_t)(rowbase + m) * ld + kb + 16 * hi;
  v16h a;
#pragma unroll
  for (int e = 0; e < 8; ++e) a[e] = r[e];
#pragma unroll
  for (int e = 0; e < 8; ++e) a[8 + e] = r[8 + e];
  return a;
}

// B fragment (32x16 f16) from a row-major [N,K] matrix (computes A * B^T):
// lane = col n (lane&15); element e -> K = kb + 16*hi + e (contiguous 16 halves).
__device__ __forceinline__ v16h load_bt(const f16* __restrict__ B, int ld,
                                        int nbase, int kb, int lane) {
  int n = nbase + (lane & 15), hi = lane >> 4;
  const f16* r = B + (size_t)n * ld + kb + 16 * hi;
  v16h b;
#pragma unroll
  for (int e = 0; e < 16; ++e) b[e] = r[e];
  return b;
}

// B fragment from LDS weight tile stored [K, NPAD] (k-major).
__device__ __forceinline__ v16h load_b_lds(const f16* __restrict__ W, int npad,
                                           int kb, int nbase, int lane) {
  int n = nbase + (lane & 15), hi = lane >> 4;
  v16h b;
#pragma unroll
  for (int e = 0; e < 16; ++e) b[e] = W[(kb + 16 * hi + e) * npad + n];
  return b;
}

// ---------------------------------------------------------------------------
// Zero-fill (used once for the K-padded q/k buffers so pad lanes stay 0).
// ---------------------------------------------------------------------------
__global__ __launch_bounds__(256) void zero_kernel(float4* __restrict__ p, size_t n) {
  size_t i = (size_t)blockIdx.x * 256 + threadIdx.x;
  if (i < n) p[i] = make_float4(0.f, 0.f, 0.f, 0.f);
}

// ---------------------------------------------------------------------------
// LayerNorm over C=64: one wave per row, 2 elems/lane, shfl reductions.
// tr=1 reads the source transposed in (i,j) -> implements swapaxes for att_end.
// ---------------------------------------------------------------------------
__global__ __launch_bounds__(256) void ln_kernel(const float* __restrict__ src,
                                                 const float* __restrict__ g,
                                                 const float* __restrict__ b,
                                                 f16* __restrict__ d16,
                                                 float* __restrict__ d32, int tr) {
  int gw   = (blockIdx.x * 256 + threadIdx.x) >> 5;
  int lane = threadIdx.x & 31;
  if (gw >= ROWS) return;
  int t = gw / (RN * RN), r1 = (gw / RN) % RN, r2 = gw % RN;
  size_t srow = tr ? (((size_t)t * RN + r2) * RN + r1) : (size_t)gw;
  const float* x = src + srow * CD;
  float v0 = x[lane], v1 = x[lane + 32];
  float s = v0 + v1;
#pragma unroll
  for (int o = 16; o; o >>= 1) s += __shfl_xor(s, o, 32);
  float mean = s * (1.f / CD);
  float d0 = v0 - mean, d1 = v1 - mean;
  float q = d0 * d0 + d1 * d1;
#pragma unroll
  for (int o = 16; o; o >>= 1) q += __shfl_xor(q, o, 32);
  float rstd = fastrsq(q * (1.f / CD) + 1e-5f);
  float y0 = d0 * rstd * g[lane] + b[lane];
  float y1 = d1 * rstd * g[lane + 32] + b[lane + 32];
  size_t orow = (size_t)gw * CD;
  if (d16) { d16[orow + lane] = (f16)y0; d16[orow + lane + 32] = (f16)y1; }
  if (d32) { d32[orow + lane] = y0;      d32[orow + lane + 32] = y1; }
}

// ---------------------------------------------------------------------------
// Fused WMMA GEMM: Y[rows,N] = f(X[rows,K] @ W1 + b1) with rich epilogues.
//  - DUAL:  Y = sigmoid(X@W1+b1) * (X@W2+b2)          (tri-mul a/b)
//  - act:   1 = sigmoid, 2 = relu
//  - mask:  per-row mask multiply; gate: * sigmoid(gate[row,col])
//  - layout 0: row-major (ostride), 1: attention K-padded [t,i,h,j,32]
//    2: channel-major [t,c,r1,r2], 3: transposed channel-major [t,c,r2,r1]
//  - resid_mode 1: z[row,col] += y ; 2: z[t,r2,r1,col] += y (att_end residual)
// 8 waves/block, 16 rows x (NT*16) cols per wave, weights staged in LDS (f16).
// Uniform-control branches are hoisted OUT of the unrolled element loops.
// ---------------------------------------------------------------------------
struct GemmP {
  const f16* X;
  const float *W1, *b1, *W2, *b2, *mask, *gate;
  float* resid;
  f16* out16;
  float* out32;
  int ldx, K, N, act, layout, resid_mode, ostride, nstore;
  float scale;
};

template <int NT, bool DUAL>
__global__ __launch_bounds__(256) void gemm_kernel(GemmP p) {
  constexpr int NPAD = NT * 16;
  __shared__ f16 sW1[128 * NPAD];
  __shared__ f16 sW2[DUAL ? 128 * NPAD : 16];
  for (int idx = threadIdx.x; idx < p.K * NPAD; idx += 256) {
    int k = idx / NPAD, n = idx % NPAD;
    sW1[idx] = (n < p.N) ? (f16)p.W1[k * p.N + n] : (f16)0.f;
    if (DUAL) sW2[idx] = (n < p.N) ? (f16)p.W2[k * p.N + n] : (f16)0.f;
  }
  __syncthreads();

  int wv = threadIdx.x >> 5, lane = threadIdx.x & 31;
  int m = lane & 15, hi = lane >> 4;
  int rowbase = blockIdx.x * 128 + wv * 16;

  v8f acc1[NT], acc2[NT];
#pragma unroll
  for (int n = 0; n < NT; ++n) { acc1[n] = (v8f){}; acc2[n] = (v8f){}; }

  for (int kb = 0; kb < p.K; kb += 32) {
    v16h a = load_a(p.X, p.ldx, rowbase, kb, lane);
#pragma unroll
    for (int nt = 0; nt < NT; ++nt) {
      v16h b = load_b_lds(sW1, NPAD, kb, nt * 16, lane);
      acc1[nt] = wmma_f16(a, b, acc1[nt]);
      if (DUAL) {
        v16h b2 = load_b_lds(sW2, NPAD, kb, nt * 16, lane);
        acc2[nt] = wmma_f16(a, b2, acc2[nt]);
      }
    }
  }

  // ---- epilogue: compute register tile first (uniform branches hoisted) ----
  float yv[NT][8];
  int rowv[8];
  int tv[8], r1v[8], r2v[8];
#pragma unroll
  for (int r = 0; r < 8; ++r) {
    int row = rowbase + r + 8 * hi;
    rowv[r] = row;
    tv[r]  = row / (RN * RN);
    r1v[r] = (row / RN) % RN;
    r2v[r] = row % RN;
  }

#pragma unroll
  for (int nt = 0; nt < NT; ++nt) {
    int col = nt * 16 + m;
    bool ok = (col < p.nstore);
    float bb1 = (ok && p.b1) ? p.b1[col] : 0.f;
    float bb2 = (DUAL && ok && p.b2) ? p.b2[col] : 0.f;
#pragma unroll
    for (int r = 0; r < 8; ++r) {
      float y = acc1[nt][r] + bb1;
      if (DUAL) {
        y = sigf(y) * (acc2[nt][r] + bb2);
      } else if (p.act == 1) {
        y = sigf(y);
      } else if (p.act == 2) {
        y = fmaxf(y, 0.f);
      }
      yv[nt][r] = y;
    }
  }
  if (p.mask) {
#pragma unroll
    for (int r = 0; r < 8; ++r) {
      float mk = p.mask[rowv[r]];
#pragma unroll
      for (int nt = 0; nt < NT; ++nt) yv[nt][r] *= mk;
    }
  }
#pragma unroll
  for (int nt = 0; nt < NT; ++nt)
#pragma unroll
    for (int r = 0; r < 8; ++r) yv[nt][r] *= p.scale;
  if (p.gate) {
#pragma unroll
    for (int nt = 0; nt < NT; ++nt) {
      int col = nt * 16 + m;
#pragma unroll
      for (int r = 0; r < 8; ++r)
        yv[nt][r] *= sigf(p.gate[(size_t)rowv[r] * CD + col]);
    }
  }

  // ---- stores (one uniform branch per mode, tight unrolled loops inside) ---
  if (p.layout == 0) {
    if (p.out16) {
#pragma unroll
      for (int nt = 0; nt < NT; ++nt) {
        int col = nt * 16 + m;
        if (col < p.nstore)
#pragma unroll
          for (int r = 0; r < 8; ++r)
            p.out16[(size_t)rowv[r] * p.ostride + col] = (f16)yv[nt][r];
      }
    }
    if (p.out32) {
#pragma unroll
      for (int nt = 0; nt < NT; ++nt) {
        int col = nt * 16 + m;
        if (col < p.nstore)
#pragma unroll
          for (int r = 0; r < 8; ++r)
            p.out32[(size_t)rowv[r] * p.ostride + col] = yv[nt][r];
      }
    }
  } else if (p.layout == 1) {
#pragma unroll
    for (int nt = 0; nt < NT; ++nt) {
      int col = nt * 16 + m;
#pragma unroll
      for (int r = 0; r < 8; ++r)
        p.out16[((((size_t)tv[r] * RN + r1v[r]) * NH + (col >> 4)) * RN + r2v[r]) * 32 +
                (col & 15)] = (f16)yv[nt][r];
    }
  } else if (p.layout == 2) {
#pragma unroll
    for (int nt = 0; nt < NT; ++nt) {
      int col = nt * 16 + m;
#pragma unroll
      for (int r = 0; r < 8; ++r)
        p.out16[(((size_t)tv[r] * CD + col) * RN + r1v[r]) * RN + r2v[r]] = (f16)yv[nt][r];
    }
  } else if (p.layout == 3) {
#pragma unroll
    for (int nt = 0; nt < NT; ++nt) {
      int col = nt * 16 + m;
#pragma unroll
      for (int r = 0; r < 8; ++r)
        p.out16[(((size_t)tv[r] * CD + col) * RN + r2v[r]) * RN + r1v[r]] = (f16)yv[nt][r];
    }
  }

  if (p.resid_mode == 1) {
#pragma unroll
    for (int nt = 0; nt < NT; ++nt) {
      int col = nt * 16 + m;
#pragma unroll
      for (int r = 0; r < 8; ++r)
        p.resid[(size_t)rowv[r] * CD + col] += yv[nt][r];
    }
  } else if (p.resid_mode == 2) {
#pragma unroll
    for (int nt = 0; nt < NT; ++nt) {
      int col = nt * 16 + m;
#pragma unroll
      for (int r = 0; r < 8; ++r)
        p.resid[(((size_t)tv[r] * RN + r2v[r]) * RN + r1v[r]) * CD + col] += yv[nt][r];
    }
  }
}

// ---------------------------------------------------------------------------
// Triangle attention (start-node form; tr handles end-node mask indexing).
// One wave per (t,i,h,q-tile16). Flash softmax over 32-wide k chunks:
//   2 WMMAs -> scores, streaming max/sum, p -> LDS -> A-frag, 1 WMMA -> ctx.
// q is pre-scaled by 1/sqrt(16) in its projection; c=16 padded to K=32.
// ---------------------------------------------------------------------------
__global__ __launch_bounds__(256) void attn_kernel(
    const f16* __restrict__ Qp, const f16* __restrict__ Kp,
    const f16* __restrict__ Vb, const float* __restrict__ gate,
    const float* __restrict__ bias, const float* __restrict__ mask,
    f16* __restrict__ out, int tr) {
  __shared__ f16 sP[8][16 * 32];
  int wv = threadIdx.x >> 5, lane = threadIdx.x & 31;
  int task = blockIdx.x * 8 + wv;           // TN*RN*NH*12 tasks, exact grid
  int ti = task / (NH * 12);
  int rem = task % (NH * 12);
  int h = rem / 12, qt = rem % 12;
  int t = ti / RN, i = ti % RN;
  int m = lane & 15, hi = lane >> 4;

  const f16* Q  = Qp + (((size_t)ti * NH + h) * RN) * 32;
  const f16* K_ = Kp + (((size_t)ti * NH + h) * RN) * 32;
  const f16* V  = Vb + (size_t)ti * RN * CD;

  v16h qa = load_a(Q, 32, qt * 16, 0, lane);
  v8f oacc = (v8f){};
  float mr[8], lr[8];
#pragma unroll
  for (int r = 0; r < 8; ++r) { mr[r] = -1e30f; lr[r] = 0.f; }
  f16* sp = &sP[wv][0];

  for (int kb = 0; kb < RN; kb += 32) {
    v16h k0 = load_bt(K_, 32, kb, 0, lane);
    v16h k1 = load_bt(K_, 32, kb + 16, 0, lane);
    v8f s0 = (v8f){}, s1 = (v8f){};
    s0 = wmma_f16(qa, k0, s0);
    s1 = wmma_f16(qa, k1, s1);

    int kk0 = kb + m, kk1 = kb + 16 + m;
    float mk0 = tr ? mask[((size_t)t * RN + kk0) * RN + i]
                   : mask[((size_t)t * RN + i) * RN + kk0];
    float mk1 = tr ? mask[((size_t)t * RN + kk1) * RN + i]
                   : mask[((size_t)t * RN + i) * RN + kk1];
    float mb0 = 1e9f * (mk0 - 1.f), mb1 = 1e9f * (mk1 - 1.f);

#pragma unroll
    for (int r = 0; r < 8; ++r) {
      int q_ = qt * 16 + r + 8 * hi;
      size_t rb = ((size_t)t * RN + q_) * RN;
      float x0 = s0[r] + mb0 + bias[(rb + kk0) * NH + h];
      float x1 = s1[r] + mb1 + bias[(rb + kk1) * NH + h];
      float sm = fmaxf(x0, x1);
      sm = fmaxf(sm, __shfl_xor(sm, 1, 32));
      sm = fmaxf(sm, __shfl_xor(sm, 2, 32));
      sm = fmaxf(sm, __shfl_xor(sm, 4, 32));
      sm = fmaxf(sm, __shfl_xor(sm, 8, 32));
      float mn = fmaxf(mr[r], sm);
      float corr = __expf(mr[r] - mn);
      float p0 = __expf(x0 - mn), p1 = __expf(x1 - mn);
      float rs = p0 + p1;
      rs += __shfl_xor(rs, 1, 32);
      rs += __shfl_xor(rs, 2, 32);
      rs += __shfl_xor(rs, 4, 32);
      rs += __shfl_xor(rs, 8, 32);
      lr[r] = lr[r] * corr + rs;
      mr[r] = mn;
      oacc[r] *= corr;
      sp[(r + 8 * hi) * 32 + m] = (f16)p0;
      sp[(r + 8 * hi) * 32 + m + 16] = (f16)p1;
    }
    __syncthreads();   // pin LDS write->read ordering (DS in-order per wave)
    v16h pa = load_a(sp, 32, 0, 0, lane);
    v16h vb;
#pragma unroll
    for (int e = 0; e < 16; ++e)
      vb[e] = V[(size_t)(kb + 16 * hi + e) * CD + h * 16 + m];
    oacc = wmma_f16(pa, vb, oacc);
  }

#pragma unroll
  for (int r = 0; r < 8; ++r) {
    int q_ = qt * 16 + r + 8 * hi;
    size_t o = ((size_t)ti * RN + q_) * CD + h * 16 + m;
    float val = oacc[r] * fastrcp(lr[r]) * sigf(gate[o]);
    out[o] = (f16)val;
  }
}

// ---------------------------------------------------------------------------
// Triangle multiplication einsum: x[t,i,j,c] = sum_k A'[i,k] * B'[j,k] per
// (t,c) batch on channel-major f16 (192x192 row-major contiguous). 1 wave per
// 16x16 output tile, 6 WMMAs (K=192).
// ---------------------------------------------------------------------------
__global__ __launch_bounds__(256) void trimul_kernel(const f16* __restrict__ A,
                                                     const f16* __restrict__ B,
                                                     float* __restrict__ X) {
  int wv = blockIdx.x * 8 + (threadIdx.x >> 5);   // TN*CD*144 tasks, exact grid
  int lane = threadIdx.x & 31;
  int m = lane & 15, hi = lane >> 4;
  int batch = wv / 144, rem = wv % 144;
  int it = rem / 12, jt = rem % 12;
  int t = batch / CD, c = batch % CD;
  const f16* Ab = A + (size_t)batch * RN * RN;
  const f16* Bb = B + (size_t)batch * RN * RN;
  v8f acc = (v8f){};
  for (int kb = 0; kb < RN; kb += 32) {
    v16h a = load_a(Ab, RN, it * 16, kb, lane);
    v16h b = load_bt(Bb, RN, jt * 16, kb, lane);
    acc = wmma_f16(a, b, acc);
  }
#pragma unroll
  for (int r = 0; r < 8; ++r) {
    int ii = it * 16 + r + 8 * hi, jj = jt * 16 + m;
    X[(((size_t)t * RN + ii) * RN + jj) * CD + c] = acc[r];
  }
}

// ---------------------------------------------------------------------------
// Host orchestration
// ---------------------------------------------------------------------------
struct AttP { const float *b_w,*g_w,*k_w,*ln_b,*ln_g,*o_b,*o_w,*q_w,*v_w; };
struct MulP { const float *a_g_b,*a_g_w,*a_p_b,*a_p_w,*b_g_b,*b_g_w,*b_p_b,*b_p_w,
              *g_b,*g_w,*ln_in_b,*ln_in_g,*ln_out_b,*ln_out_g,*z_b,*z_w; };
struct TrP  { const float *b1,*b2,*ln_b,*ln_g,*w1,*w2; };
struct BlockP { AttP att_end, att_start; MulP mul_in, mul_out; TrP trans; };

static inline void launch_gemm(const GemmP& gp, int nt, bool dual, hipStream_t s) {
  dim3 grid(ROWS / 128), block(256);
  if (dual)          gemm_kernel<4, true><<<grid, block, 0, s>>>(gp);
  else if (nt == 1)  gemm_kernel<1, false><<<grid, block, 0, s>>>(gp);
  else if (nt == 8)  gemm_kernel<8, false><<<grid, block, 0, s>>>(gp);
  else               gemm_kernel<4, false><<<grid, block, 0, s>>>(gp);
}

extern "C" void kernel_launch(void* const* d_in, const int* in_sizes, int n_in,
                              void* d_out, int out_size, void* d_ws, size_t ws_size,
                              hipStream_t stream) {
  (void)in_sizes; (void)n_in; (void)out_size; (void)ws_size;
  const float* t_in = (const float*)d_in[0];
  const float* mask = (const float*)d_in[1];

  // params flattened in jax pytree (sorted-key) order
  int ix = 2;
  auto nx = [&]() { return (const float*)d_in[ix++]; };
  auto rdAtt = [&](AttP& a) { a.b_w=nx(); a.g_w=nx(); a.k_w=nx(); a.ln_b=nx();
                              a.ln_g=nx(); a.o_b=nx(); a.o_w=nx(); a.q_w=nx(); a.v_w=nx(); };
  auto rdMul = [&](MulP& p) { p.a_g_b=nx(); p.a_g_w=nx(); p.a_p_b=nx(); p.a_p_w=nx();
                              p.b_g_b=nx(); p.b_g_w=nx(); p.b_p_b=nx(); p.b_p_w=nx();
                              p.g_b=nx(); p.g_w=nx(); p.ln_in_b=nx(); p.ln_in_g=nx();
                              p.ln_out_b=nx(); p.ln_out_g=nx(); p.z_b=nx(); p.z_w=nx(); };
  auto rdTr  = [&](TrP& p)  { p.b1=nx(); p.b2=nx(); p.ln_b=nx(); p.ln_g=nx();
                              p.w1=nx(); p.w2=nx(); };
  BlockP blk[2];
  for (int b = 0; b < 2; ++b) {
    rdAtt(blk[b].att_end); rdAtt(blk[b].att_start);
    rdMul(blk[b].mul_in);  rdMul(blk[b].mul_out);  rdTr(blk[b].trans);
  }
  const float* fin_b = nx();
  const float* fin_g = nx();

  // workspace carve
  uintptr_t w = (uintptr_t)d_ws;
  auto carve = [&](size_t bytes) -> void* {
    uintptr_t r = (w + 255) & ~(uintptr_t)255; w = r + bytes; return (void*)r;
  };
  float* z    = (float*)carve((size_t)ROWS * CD * 4);
  f16*   zl   = (f16*)  carve((size_t)ROWS * CD * 2);
  f16*   qpad = (f16*)  carve((size_t)TN * RN * NH * RN * 32 * 2);
  f16*   kpad = (f16*)  carve((size_t)TN * RN * NH * RN * 32 * 2);
  f16*   vbuf = (f16*)  carve((size_t)ROWS * CD * 2);
  float* gbuf = (float*)carve((size_t)ROWS * CD * 4);
  float* bbuf = (float*)carve((size_t)ROWS * NH * 4);
  f16*   obuf = (f16*)  carve((size_t)ROWS * CD * 2);
  f16*   acm  = (f16*)  carve((size_t)TN * CD * RN * RN * 2);
  f16*   bcm  = (f16*)  carve((size_t)TN * CD * RN * RN * 2);
  float* xbuf = (float*)carve((size_t)ROWS * CD * 4);
  f16*   hbuf = (f16*)  carve((size_t)ROWS * 128 * 2);

  auto base = [&](const f16* X) {
    GemmP g{}; g.X = X; g.ldx = CD; g.K = CD; g.N = CD;
    g.ostride = CD; g.nstore = CD; g.scale = 1.f; return g;
  };
  auto LN = [&](const float* s, const float* gg, const float* bb,
                f16* d16, float* d32, int tr) {
    ln_kernel<<<ROWS / 8, 256, 0, stream>>>(s, gg, bb, d16, d32, tr);
  };

  auto runAtt = [&](const AttP& a, int tr) {
    LN(z, a.ln_g, a.ln_b, zl, nullptr, tr);
    { GemmP g = base(zl); g.W1 = a.q_w; g.scale = 0.25f; g.layout = 1;
      g.out16 = qpad; launch_gemm(g, 4, false, stream); }
    { GemmP g = base(zl); g.W1 = a.k_w; g.layout = 1; g.out16 = kpad;
      launch_gemm(g, 4, false, stream); }
    { GemmP g = base(zl); g.W1 = a.v_w; g.out16 = vbuf;
      launch_gemm(g, 4, false, stream); }
    { GemmP g = base(zl); g.W1 = a.g_w; g.out32 = gbuf;
      launch_gemm(g, 4, false, stream); }
    { GemmP g = base(zl); g.W1 = a.b_w; g.N = NH; g.nstore = NH; g.ostride = NH;
      g.out32 = bbuf; launch_gemm(g, 1, false, stream); }
    attn_kernel<<<(TN * RN * NH * 12) / 8, 256, 0, stream>>>(
        qpad, kpad, vbuf, gbuf, bbuf, mask, obuf, tr);
    { GemmP g = base(obuf); g.W1 = a.o_w; g.b1 = a.o_b; g.resid = z;
      g.resid_mode = tr ? 2 : 1; launch_gemm(g, 4, false, stream); }
  };

  auto runMul = [&](const MulP& mp, int outg) {
    LN(z, mp.ln_in_g, mp.ln_in_b, zl, nullptr, 0);
    { GemmP g = base(zl); g.W1 = mp.a_g_w; g.b1 = mp.a_g_b; g.W2 = mp.a_p_w;
      g.b2 = mp.a_p_b; g.mask = mask; g.layout = outg ? 2 : 3; g.out16 = acm;
      launch_gemm(g, 4, true, stream); }
    { GemmP g = base(zl); g.W1 = mp.b_g_w; g.b1 = mp.b_g_b; g.W2 = mp.b_p_w;
      g.b2 = mp.b_p_b; g.mask = mask; g.layout = outg ? 2 : 3; g.out16 = bcm;
      launch_gemm(g, 4, true, stream); }
    { GemmP g = base(zl); g.W1 = mp.g_w; g.b1 = mp.g_b; g.out32 = gbuf;
      launch_gemm(g, 4, false, stream); }
    trimul_kernel<<<(TN * CD * 144) / 8, 256, 0, stream>>>(acm, bcm, xbuf);
    LN(xbuf, mp.ln_out_g, mp.ln_out_b, zl, nullptr, 0);
    { GemmP g = base(zl); g.W1 = mp.z_w; g.b1 = mp.z_b; g.gate = gbuf;
      g.resid = z; g.resid_mode = 1; launch_gemm(g, 4, false, stream); }
  };

  // z = t ; zero the K-padded q/k buffers once (pad region stays zero)
  hipMemcpyAsync(z, t_in, (size_t)ROWS * CD * 4, hipMemcpyDeviceToDevice, stream);
  size_t qk4 = (size_t)TN * RN * NH * RN * 32 * 2 / 16;
  zero_kernel<<<(unsigned)((qk4 + 255) / 256), 256, 0, stream>>>((float4*)qpad, qk4);
  zero_kernel<<<(unsigned)((qk4 + 255) / 256), 256, 0, stream>>>((float4*)kpad, qk4);

  for (int b = 0; b < 2; ++b) {
    runAtt(blk[b].att_start, 0);
    runAtt(blk[b].att_end, 1);
    runMul(blk[b].mul_out, 1);
    runMul(blk[b].mul_in, 0);
    // transition
    const TrP& tp = blk[b].trans;
    LN(z, tp.ln_g, tp.ln_b, zl, nullptr, 0);
    { GemmP g = base(zl); g.W1 = tp.w1; g.b1 = tp.b1; g.N = 128; g.nstore = 128;
      g.ostride = 128; g.act = 2; g.out16 = hbuf; launch_gemm(g, 8, false, stream); }
    { GemmP g = base(hbuf); g.ldx = 128; g.K = 128; g.W1 = tp.w2; g.b1 = tp.b2;
      g.mask = mask; g.resid = z; g.resid_mode = 1; launch_gemm(g, 4, false, stream); }
  }
  LN(z, fin_g, fin_b, nullptr, (float*)d_out, 0);
}
